// NeuralBellmanFordNetwork_11003706213174
// MI455X (gfx1250) — matrix operand order repaired and checked by reading, not compiled, not run on hardware
//
#include <hip/hip_runtime.h>

// ---------------- problem constants (match reference) ----------------
#define N_NODE 20000
#define BQ     8        // batch
#define DIM    32       // D
#define NREL   100      // R2
#define NLAYER 6        // L
#define KTOT   416      // 13*D
#define KT     13       // K tiles of 32
#define BD     (BQ*DIM) // 256 floats per node row
#define NNEG   32

// ---------------- CDNA5 WMMA types ----------------
typedef __attribute__((ext_vector_type(16))) __bf16 v16bf;
typedef __attribute__((ext_vector_type(8)))  float  v8f;

struct alignas(16) BFrag { uint4 a, b; };           // 32B: 16 bf16 (B-matrix per-lane frag)
struct AFrag { unsigned int u[8]; };                // 32B: 16 bf16 (A-matrix per-lane frag)

__device__ __forceinline__ unsigned short f2bf(float f) {
  unsigned int u = __float_as_uint(f);
  unsigned int r = u + 0x7FFFu + ((u >> 16) & 1u);  // round-to-nearest-even
  return (unsigned short)(r >> 16);
}

// ---------------- one-time setup kernels ----------------
__global__ void k_zero(int* cnt, int* cursor, float* scalesum) {
  int i = blockIdx.x * blockDim.x + threadIdx.x;
  if (i < N_NODE) { cnt[i] = 0; cursor[i] = 0; }
  if (i == 0) *scalesum = 0.f;
}

__global__ void k_count(const int* __restrict__ edge_dst, int* __restrict__ cnt, int E) {
  int e = blockIdx.x * blockDim.x + threadIdx.x;
  if (e < E) atomicAdd(&cnt[edge_dst[e]], 1);
}

// single-block exclusive scan of cnt -> csr_off, plus deg = cnt+1
__global__ __launch_bounds__(1024) void k_scan(const int* __restrict__ cnt,
                                               int* __restrict__ csr_off,
                                               float* __restrict__ deg) {
  __shared__ int part[1024];
  int tid = threadIdx.x;
  const int chunk = (N_NODE + 1023) / 1024;
  int beg = tid * chunk;
  int end = beg + chunk; if (end > N_NODE) end = N_NODE;
  int s = 0;
  for (int i = beg; i < end; ++i) s += cnt[i];
  part[tid] = s;
  __syncthreads();
  for (int off = 1; off < 1024; off <<= 1) {
    int v = (tid >= off) ? part[tid - off] : 0;
    __syncthreads();
    if (tid >= off) part[tid] += v;
    __syncthreads();
  }
  int excl = (tid == 0) ? 0 : part[tid - 1];
  for (int i = beg; i < end; ++i) {
    csr_off[i] = excl;
    excl += cnt[i];
    deg[i] = (float)(cnt[i] + 1);
  }
  if (tid == 1023) csr_off[N_NODE] = part[1023];
}

__global__ void k_scatter(const int* __restrict__ edge_src, const int* __restrict__ edge_dst,
                          const int* __restrict__ edge_type, const int* __restrict__ csr_off,
                          int* __restrict__ cursor, int* __restrict__ csr_src,
                          int* __restrict__ csr_type, int E) {
  int e = blockIdx.x * blockDim.x + threadIdx.x;
  if (e >= E) return;
  int d = edge_dst[e];
  int p = csr_off[d] + atomicAdd(&cursor[d], 1);
  csr_src[p]  = edge_src[e];
  csr_type[p] = edge_type[e];
}

__global__ void k_scale_reduce(const float* __restrict__ deg, float* __restrict__ scalesum) {
  __shared__ float red[256];
  int i = blockIdx.x * blockDim.x + threadIdx.x;
  float l = (i < N_NODE) ? __logf(deg[i]) : 0.f;
  red[threadIdx.x] = l;
  __syncthreads();
  for (int o = 128; o > 0; o >>= 1) {
    if (threadIdx.x < o) red[threadIdx.x] += red[threadIdx.x + o];
    __syncthreads();
  }
  if (threadIdx.x == 0) atomicAdd(scalesum, red[0]);
}

__global__ void k_scales(const float* __restrict__ deg, const float* __restrict__ scalesum,
                         float* __restrict__ scales) {
  int i = blockIdx.x * blockDim.x + threadIdx.x;
  if (i >= N_NODE) return;
  float mean = *scalesum / (float)N_NODE;
  float s = __logf(deg[i]) / mean;
  scales[i * 3 + 0] = 1.f;
  scales[i * 3 + 1] = s;
  scales[i * 3 + 2] = 1.f / fmaxf(s, 0.01f);
}

__global__ void k_query(const int* __restrict__ r_index, const float* __restrict__ query_emb,
                        float* __restrict__ query) {
  int t = threadIdx.x;             // 256 = B*D
  int b = t >> 5, d = t & 31;
  query[t] = query_emb[r_index[b] * DIM + d];
}

__global__ void k_xinit(const int* __restrict__ h_index, const float* __restrict__ query,
                        float* __restrict__ x0) {
  int idx = blockIdx.x * blockDim.x + threadIdx.x;   // N*B*D
  int n = idx >> 8;
  int t = idx & 255;
  int b = t >> 5, d = t & 31;
  x0[idx] = (h_index[b] == n) ? query[b * DIM + d] : 0.f;
}

// pre-swizzle Wl (L,32,416) into WMMA B-matrix per-lane bf16 fragments:
// Wb[layer][tile][lane][e], B 32x16: lane -> N=lane&15, K = (lane>=16?16:0)+2v+h
__global__ void k_wb(const float* __restrict__ Wl, unsigned short* __restrict__ Wb) {
  int idx = blockIdx.x * blockDim.x + threadIdx.x;   // 6*13*32*16
  if (idx >= NLAYER * KT * 32 * 16) return;
  int t = idx;
  int e    = t & 15;  t >>= 4;
  int lane = t & 31;  t >>= 5;
  int tile = t % KT;
  int layer = t / KT;
  int n   = lane & 15;
  int grp = lane >> 4;
  int v = e >> 1, h = e & 1;
  int k = tile * 32 + grp * 16 + v * 2 + h;
  float val = Wl[((size_t)layer * DIM + n) * KTOT + k];
  Wb[idx] = f2bf(val);
}

// per-layer relation embedding: rel[b][r][d] = dot(query[b], Wr_l[r*D+d,:]) + br_l[r*D+d]
__global__ void k_rel(const float* __restrict__ query, const float* __restrict__ Wr_l,
                      const float* __restrict__ br_l, float* __restrict__ rel) {
  int idx = blockIdx.x * blockDim.x + threadIdx.x;   // B*R2*D = 25600
  if (idx >= BQ * NREL * DIM) return;
  int j = idx % (NREL * DIM);
  int b = idx / (NREL * DIM);
  float acc = br_l[j];
  const float* w = Wr_l + (size_t)j * DIM;
  const float* q = query + b * DIM;
#pragma unroll
  for (int k = 0; k < DIM; ++k) acc = fmaf(q[k], w[k], acc);
  rel[idx] = acc;
}

// ---------------- fused aggregation + WMMA GEMM layer kernel ----------------
// block = 256 threads (8 waves), owns 8 nodes -> 64 rows of the (M=N*B, K=416) A panel
__global__ __launch_bounds__(256) void k_layer(
    const float* __restrict__ x_in, float* __restrict__ x_out,
    const int* __restrict__ csr_off, const int* __restrict__ csr_src,
    const int* __restrict__ csr_type, const float* __restrict__ rel,
    const float* __restrict__ scales, const float* __restrict__ deg,
    const int* __restrict__ h_index, const float* __restrict__ query,
    const unsigned short* __restrict__ Wb_l, const float* __restrict__ bl_l) {
  __shared__ unsigned short Apan[64][KTOT + 8];   // bf16 A panel, padded stride

  const int tid = threadIdx.x;
  const int b = tid >> 5;          // 0..7 batch
  const int d = tid & 31;          // 0..31 dim
  const int n0 = blockIdx.x * 8;
  const int hb = h_index[b];
  const float qv = query[b * DIM + d];

  // ---- phase 1: aggregate stats per node, emit bf16 A rows into LDS ----
  for (int nl = 0; nl < 8; ++nl) {
    const int n = n0 + nl;
    float bval = (hb == n) ? qv : 0.f;
    float s = bval, s2 = bval * bval, mx = bval, mn = bval;
    const int e0 = csr_off[n], e1 = csr_off[n + 1];
    for (int e = e0; e < e1; ++e) {
      int src = csr_src[e];
      int et  = csr_type[e];
      float v = x_in[(size_t)src * BD + tid] * rel[b * (NREL * DIM) + et * DIM + d];
      s += v; s2 += v * v;
      mx = fmaxf(mx, v); mn = fminf(mn, v);
    }
    float dg   = deg[n];
    float mean = s / dg;
    float sqm  = s2 / dg;
    float stdv = __fsqrt_rn(fmaxf(sqm - mean * mean, 1e-6f));
    float sc1 = scales[n * 3 + 1];
    float sc2 = scales[n * 3 + 2];
    int row = nl * BQ + b;
    Apan[row][d] = f2bf(x_in[(size_t)n * BD + tid]);     // x part, k in [0,32)
    float st[4] = {mean, mx, mn, stdv};
#pragma unroll
    for (int t = 0; t < 4; ++t) {
      int base = DIM + (d * 4 + t) * 3;                  // (d*4+stat)*3 + sc ordering
      Apan[row][base + 0] = f2bf(st[t]);
      Apan[row][base + 1] = f2bf(st[t] * sc1);
      Apan[row][base + 2] = f2bf(st[t] * sc2);
    }
  }
  __syncthreads();

  // ---- phase 2: 8 waves x (16 rows x 16 cols) WMMA tiles, K = 13 x 32 ----
  const int wave = tid >> 5;
  const int lane = tid & 31;
  const int rt = wave >> 1;        // row tile 0..3
  const int ct = wave & 1;         // col tile 0..1
  const int m   = lane & 15;
  const int grp = lane >> 4;
  const int arow = rt * 16 + m;

  v8f c = {};
#pragma unroll
  for (int tk = 0; tk < KT; ++tk) {
    // B fragment: one 32B per-lane load from pre-swizzled weights
    BFrag bf = *(const BFrag*)(Wb_l + ((size_t)tk * 32 + lane) * 16);
    v16bf bmat = __builtin_bit_cast(v16bf, bf);
    // A fragment from LDS: 16-bit A 16x32 layout
    AFrag af;
#pragma unroll
    for (int v = 0; v < 8; ++v) {
      int k0 = tk * 32 + grp * 16 + v * 2;
      unsigned int lo = Apan[arow][k0];
      unsigned int hi = Apan[arow][k0 + 1];
      af.u[v] = lo | (hi << 16);
    }
    v16bf amat = __builtin_bit_cast(v16bf, af);
    c = __builtin_amdgcn_wmma_f32_16x16x32_bf16(false, amat, false, bmat,
                                                (short)0, c, false, false);
  }

  // ---- phase 3: bias + relu epilogue, scatter to x_out ----
#pragma unroll
  for (int j = 0; j < 8; ++j) {
    int mm  = j + grp * 8;
    int row = rt * 16 + mm;
    int nl  = row >> 3;
    int bb  = row & 7;
    int nout = (lane & 15) + ct * 16;
    float val = fmaxf(c[j] + bl_l[nout], 0.f);
    x_out[(size_t)(n0 + nl) * BD + bb * DIM + nout] = val;
  }
}

// ---------------- final scoring MLP ----------------
__global__ __launch_bounds__(256) void k_score(
    const float* __restrict__ x_fin, const float* __restrict__ query,
    const int* __restrict__ t_index, const float* __restrict__ W1,
    const float* __restrict__ b1, const float* __restrict__ W2,
    const float* __restrict__ b2, float* __restrict__ out) {
  __shared__ float sW1[64 * 64];
  __shared__ float sb1[64], sW2[64];
  int tid = threadIdx.x;
  for (int i = tid; i < 64 * 64; i += 256) sW1[i] = W1[i];
  if (tid < 64) { sb1[tid] = b1[tid]; sW2[tid] = W2[tid]; }
  __syncthreads();

  int b = tid >> 5, neg = tid & 31;           // 256 = B * NEG
  int t = t_index[b * NNEG + neg];
  float f[64];
#pragma unroll
  for (int k = 0; k < DIM; ++k) f[k] = x_fin[(size_t)t * BD + b * DIM + k];
#pragma unroll
  for (int k = 0; k < DIM; ++k) f[DIM + k] = query[b * DIM + k];
  float score = b2[0];
  for (int j = 0; j < 64; ++j) {
    float h = sb1[j];
#pragma unroll
    for (int k = 0; k < 64; ++k) h = fmaf(sW1[j * 64 + k], f[k], h);
    score = fmaf(sW2[j], fmaxf(h, 0.f), score);
  }
  out[b * NNEG + neg] = score;
}

// ---------------- host launch ----------------
extern "C" void kernel_launch(void* const* d_in, const int* in_sizes, int n_in,
                              void* d_out, int out_size, void* d_ws, size_t ws_size,
                              hipStream_t stream) {
  const int*   edge_src  = (const int*)d_in[0];
  const int*   edge_dst  = (const int*)d_in[1];
  const int*   edge_type = (const int*)d_in[2];
  const int*   h_index   = (const int*)d_in[3];
  const int*   t_index   = (const int*)d_in[4];
  const int*   r_index   = (const int*)d_in[5];
  const float* query_emb = (const float*)d_in[6];
  const float* Wr        = (const float*)d_in[7];
  const float* br        = (const float*)d_in[8];
  const float* Wl        = (const float*)d_in[9];
  const float* bl        = (const float*)d_in[10];
  const float* W1        = (const float*)d_in[11];
  const float* b1        = (const float*)d_in[12];
  const float* W2        = (const float*)d_in[13];
  const float* b2        = (const float*)d_in[14];
  const int E = in_sizes[0];

  char* ws = (char*)d_ws;
  size_t off = 0;
  auto alloc = [&](size_t bytes) -> char* {
    char* p = ws + off;
    off = (off + bytes + 255) & ~(size_t)255;
    return p;
  };
  int*   cnt      = (int*)alloc((size_t)N_NODE * 4);
  int*   cursor   = (int*)alloc((size_t)N_NODE * 4);
  int*   csr_off  = (int*)alloc((size_t)(N_NODE + 1) * 4);
  int*   csr_src  = (int*)alloc((size_t)E * 4);
  int*   csr_type = (int*)alloc((size_t)E * 4);
  float* deg      = (float*)alloc((size_t)N_NODE * 4);
  float* scales   = (float*)alloc((size_t)N_NODE * 3 * 4);
  float* query    = (float*)alloc((size_t)BD * 4);
  float* rel      = (float*)alloc((size_t)BQ * NREL * DIM * 4);
  float* scalesum = (float*)alloc(4);
  unsigned short* Wb = (unsigned short*)alloc((size_t)NLAYER * KT * 32 * 16 * 2);
  float* x0 = (float*)alloc((size_t)N_NODE * BD * 4);
  float* x1 = (float*)alloc((size_t)N_NODE * BD * 4);
  (void)ws_size; (void)n_in; (void)out_size;

  const int TB = 256;
  int gN = (N_NODE + TB - 1) / TB;
  int gE = (E + TB - 1) / TB;

  k_zero<<<gN, TB, 0, stream>>>(cnt, cursor, scalesum);
  k_count<<<gE, TB, 0, stream>>>(edge_dst, cnt, E);
  k_scan<<<1, 1024, 0, stream>>>(cnt, csr_off, deg);
  k_scatter<<<gE, TB, 0, stream>>>(edge_src, edge_dst, edge_type, csr_off,
                                   cursor, csr_src, csr_type, E);
  k_scale_reduce<<<gN, TB, 0, stream>>>(deg, scalesum);
  k_scales<<<gN, TB, 0, stream>>>(deg, scalesum, scales);
  k_query<<<1, TB, 0, stream>>>(r_index, query_emb, query);
  k_xinit<<<(N_NODE * BD) / TB, TB, 0, stream>>>(h_index, query, x0);
  k_wb<<<(NLAYER * KT * 32 * 16 + TB - 1) / TB, TB, 0, stream>>>(Wl, Wb);

  float* xbuf[2] = {x0, x1};
  for (int l = 0; l < NLAYER; ++l) {
    k_rel<<<(BQ * NREL * DIM + TB - 1) / TB, TB, 0, stream>>>(
        query, Wr + (size_t)l * NREL * DIM * DIM, br + (size_t)l * NREL * DIM, rel);
    k_layer<<<N_NODE / 8, TB, 0, stream>>>(
        xbuf[l & 1], xbuf[(l + 1) & 1], csr_off, csr_src, csr_type, rel,
        scales, deg, h_index, query, Wb + (size_t)l * KT * 32 * 16,
        bl + (size_t)l * DIM);
  }
  k_score<<<1, TB, 0, stream>>>(xbuf[NLAYER & 1], query, t_index, W1, b1, W2, b2,
                                (float*)d_out);
}
